// DynamicCategoryMoE_58987080843589
// MI455X (gfx1250) — matrix operand order repaired and checked by reading, not compiled, not run on hardware
//
#include <hip/hip_runtime.h>

// Problem dims
#define TKN  2048   // B*S tokens
#define HDIM 1024
#define IDIM 4096
#define EDIM 8

// Block tile: 128 (M) x 128 (N), K-step 32. 8 waves, each owns 32M x 64N.
#define AS 40   // A LDS row pitch in bf16 elems (80B: 16B aligned, conflict-free)
#define BS 40   // B LDS row pitch in bf16 elems

typedef __bf16 v2bf  __attribute__((ext_vector_type(2)));
typedef __bf16 v8bf  __attribute__((ext_vector_type(8)));
typedef __bf16 v16bf __attribute__((ext_vector_type(16)));
typedef float  v8f   __attribute__((ext_vector_type(8)));

__device__ __forceinline__ unsigned pk_bf2(float a, float b) {
  v2bf p; p[0] = (__bf16)a; p[1] = (__bf16)b;   // fuses to v_cvt_pk_bf16_f32
  return __builtin_bit_cast(unsigned, p);
}

__device__ __forceinline__ float gelu_exact(float x) {
  return 0.5f * x * (1.0f + erff(x * 0.70710678118654752f));
}

__device__ __forceinline__ v16bf cat16(v8bf lo, v8bf hi) {
  return __builtin_shufflevector(lo, hi, 0,1,2,3,4,5,6,7,8,9,10,11,12,13,14,15);
}

__device__ __forceinline__ unsigned lds_off32(const void* p) {
  // LDS aperture: addr[31:0] is the in-LDS byte offset
  return (unsigned)(unsigned long long)p;
}

// ---------------------------------------------------------------- router ----
__global__ __launch_bounds__(256)
void k_init(int* __restrict__ counts, int* __restrict__ cursors) {
  if (threadIdx.x < EDIM) { counts[threadIdx.x] = 0; cursors[threadIdx.x] = 0; }
}

__global__ __launch_bounds__(256)
void k_router(const float* __restrict__ x, const float* __restrict__ rw,
              const float* __restrict__ rb, int* __restrict__ eidx,
              int* __restrict__ counts) {
  const int wid = threadIdx.x >> 5, lane = threadIdx.x & 31;
  const int t = blockIdx.x * 8 + wid;
  float s[EDIM];
#pragma unroll
  for (int e = 0; e < EDIM; ++e) s[e] = 0.0f;
  for (int k = lane; k < HDIM; k += 32) {
    float xv = x[(size_t)t * HDIM + k];
    const float4* r4 = (const float4*)(rw + (size_t)k * EDIM);
    float4 a = r4[0], b = r4[1];
    s[0] = fmaf(xv, a.x, s[0]); s[1] = fmaf(xv, a.y, s[1]);
    s[2] = fmaf(xv, a.z, s[2]); s[3] = fmaf(xv, a.w, s[3]);
    s[4] = fmaf(xv, b.x, s[4]); s[5] = fmaf(xv, b.y, s[5]);
    s[6] = fmaf(xv, b.z, s[6]); s[7] = fmaf(xv, b.w, s[7]);
  }
#pragma unroll
  for (int m = 16; m >= 1; m >>= 1) {
#pragma unroll
    for (int e = 0; e < EDIM; ++e) s[e] += __shfl_xor(s[e], m, 32);
  }
  if (lane == 0) {
    int best = 0; float bv = s[0] + rb[0];
#pragma unroll
    for (int e = 1; e < EDIM; ++e) {
      float v = s[e] + rb[e];
      if (v > bv) { bv = v; best = e; }   // strict > == argmax first-max
    }
    eidx[t] = best;
    atomicAdd(&counts[best], 1);
  }
}

__global__ void k_scan(const int* __restrict__ counts, int* __restrict__ offsets) {
  if (threadIdx.x == 0) {
    int a = 0;
    for (int e = 0; e < EDIM; ++e) { offsets[e] = a; a += counts[e]; }
  }
}

__global__ __launch_bounds__(256)
void k_fill(const int* __restrict__ eidx, const int* __restrict__ offsets,
            int* __restrict__ cursors, int* __restrict__ list) {
  int t = blockIdx.x * 256 + threadIdx.x;
  if (t < TKN) {
    int e = eidx[t];
    int p = atomicAdd(&cursors[e], 1);
    list[offsets[e] + p] = t;
  }
}

// ------------------------------------------------------------ FFN layer 1 ---
// h[compact_row, i] = gelu((x[tok]+0.1*emb[e]) . w1[e][:,i] + b1[e][i]) -> bf16
__global__ __launch_bounds__(256)
void k_ffn1(const float* __restrict__ x, const float* __restrict__ w1,
            const float* __restrict__ b1, const float* __restrict__ emb,
            const int* __restrict__ list, const int* __restrict__ counts,
            const int* __restrict__ offsets, unsigned short* __restrict__ hbuf) {
  const int e  = blockIdx.z;
  const int mt = blockIdx.y;
  const int nt = blockIdx.x;
  const int cnt = counts[e];
  if (mt * 128 >= cnt) return;
  const int off = offsets[e];
  const int n0 = nt * 128;

  __shared__ __align__(16) unsigned short Asm[128 * AS];
  __shared__ __align__(16) unsigned short Bsm[128 * BS];

  const int tid = threadIdx.x;
  const int lane = tid & 31, wid = tid >> 5;
  const int wm = wid >> 1, wn = wid & 1;     // waves: 4 (M) x 2 (N) over 128x128
  const int l16 = lane & 15, hb = lane >> 4;

  const float* wbase = w1 + (size_t)e * HDIM * IDIM;
  const float* ebase = emb + (size_t)e * HDIM;

  // A staging: thread -> (row, 16-elem K chunk)
  const int arow = tid >> 1, ach = (tid & 1) * 16;
  {
    // nothing
  }
  int arl = mt * 128 + arow;
  const int atok = list[off + (arl < cnt ? arl : cnt - 1)];   // hoisted gather idx
  const float* xrow = x + (size_t)atok * HDIM;

  // B staging: thread -> (K quad, 4 N cols)
  const int bk = (tid >> 5) * 4;          // 0,4,...,28
  const int bn = (tid & 31) * 4;          // 0..124

  v8f acc[2][4];
#pragma unroll
  for (int i = 0; i < 2; ++i)
#pragma unroll
    for (int j = 0; j < 4; ++j) acc[i][j] = (v8f){};

  for (int k0 = 0; k0 < HDIM; k0 += 32) {
    __syncthreads();
    { // stage A: (x[tok] + 0.1*emb) -> bf16 [M][K]
      const float4* xp = (const float4*)(xrow + k0 + ach);
      const float4* ep = (const float4*)(ebase + k0 + ach);
      float4 xa = xp[0], xb = xp[1], xc = xp[2], xd = xp[3];
      float4 ea = ep[0], eb = ep[1], ec = ep[2], ed = ep[3];
      uint4 lo, hi;
      lo.x = pk_bf2(fmaf(0.1f, ea.x, xa.x), fmaf(0.1f, ea.y, xa.y));
      lo.y = pk_bf2(fmaf(0.1f, ea.z, xa.z), fmaf(0.1f, ea.w, xa.w));
      lo.z = pk_bf2(fmaf(0.1f, eb.x, xb.x), fmaf(0.1f, eb.y, xb.y));
      lo.w = pk_bf2(fmaf(0.1f, eb.z, xb.z), fmaf(0.1f, eb.w, xb.w));
      hi.x = pk_bf2(fmaf(0.1f, ec.x, xc.x), fmaf(0.1f, ec.y, xc.y));
      hi.y = pk_bf2(fmaf(0.1f, ec.z, xc.z), fmaf(0.1f, ec.w, xc.w));
      hi.z = pk_bf2(fmaf(0.1f, ed.x, xd.x), fmaf(0.1f, ed.y, xd.y));
      hi.w = pk_bf2(fmaf(0.1f, ed.z, xd.z), fmaf(0.1f, ed.w, xd.w));
      *(uint4*)&Asm[arow * AS + ach]     = lo;
      *(uint4*)&Asm[arow * AS + ach + 8] = hi;
    }
    { // stage B: 4 w1 k-rows x 4 n-cols, pack K-quads -> bf16 transposed [N][K]
      const float* r0 = wbase + (size_t)(k0 + bk) * IDIM + n0 + bn;
      float4 p0 = *(const float4*)(r0);
      float4 p1 = *(const float4*)(r0 + IDIM);
      float4 p2 = *(const float4*)(r0 + 2 * IDIM);
      float4 p3 = *(const float4*)(r0 + 3 * IDIM);
      uint2 q;
      q.x = pk_bf2(p0.x, p1.x); q.y = pk_bf2(p2.x, p3.x);
      *(uint2*)&Bsm[(bn + 0) * BS + bk] = q;
      q.x = pk_bf2(p0.y, p1.y); q.y = pk_bf2(p2.y, p3.y);
      *(uint2*)&Bsm[(bn + 1) * BS + bk] = q;
      q.x = pk_bf2(p0.z, p1.z); q.y = pk_bf2(p2.z, p3.z);
      *(uint2*)&Bsm[(bn + 2) * BS + bk] = q;
      q.x = pk_bf2(p0.w, p1.w); q.y = pk_bf2(p2.w, p3.w);
      *(uint2*)&Bsm[(bn + 3) * BS + bk] = q;
      if (k0 + 32 < HDIM)
        __builtin_prefetch(r0 + (size_t)32 * IDIM, 0, 0);
    }
    __syncthreads();
    // fragments per CDNA5 wave32 layouts
    const unsigned short* pa = &Asm[(wm * 32 + l16) * AS + hb * 8];
    v16bf a0 = cat16(*(const v8bf*)pa, *(const v8bf*)(pa + 16));
    v16bf a1 = cat16(*(const v8bf*)(pa + 16 * AS), *(const v8bf*)(pa + 16 * AS + 16));
    const unsigned short* pb = &Bsm[(wn * 64 + l16) * BS + hb * 8];
    v16bf bf[4];
#pragma unroll
    for (int j = 0; j < 4; ++j)
      bf[j] = cat16(*(const v8bf*)(pb + j * 16 * BS), *(const v8bf*)(pb + j * 16 * BS + 16));
#pragma unroll
    for (int j = 0; j < 4; ++j) {
      acc[0][j] = __builtin_amdgcn_wmma_f32_16x16x32_bf16(false, a0, false, bf[j], (short)0, acc[0][j], false, false);
      acc[1][j] = __builtin_amdgcn_wmma_f32_16x16x32_bf16(false, a1, false, bf[j], (short)0, acc[1][j], false, false);
    }
  }

#pragma unroll
  for (int j = 0; j < 4; ++j) {
    const int nc = n0 + wn * 64 + j * 16 + l16;
    const float bias = b1[(size_t)e * IDIM + nc];
#pragma unroll
    for (int i = 0; i < 2; ++i) {
#pragma unroll
      for (int r = 0; r < 8; ++r) {
        int m = wm * 32 + i * 16 + r + hb * 8;
        if (mt * 128 + m < cnt) {
          size_t row = (size_t)(off + mt * 128 + m) * IDIM;
          hbuf[row + nc] =
              __builtin_bit_cast(unsigned short, (__bf16)gelu_exact(acc[i][j][r] + bias));
        }
      }
    }
  }
}

// ------------------------------------------------------------ FFN layer 2 ---
// out[tok, h] = h_compact . w2[e][:,h] + b2[e][h]
__global__ __launch_bounds__(256)
void k_ffn2(const unsigned short* __restrict__ hbuf, const float* __restrict__ w2,
            const float* __restrict__ b2, const int* __restrict__ list,
            const int* __restrict__ counts, const int* __restrict__ offsets,
            float* __restrict__ out) {
  const int e  = blockIdx.z;
  const int mt = blockIdx.y;
  const int nt = blockIdx.x;
  const int cnt = counts[e];
  if (mt * 128 >= cnt) return;
  const int off = offsets[e];
  const int n0 = nt * 128;

  __shared__ __align__(16) unsigned short Asm[128 * AS];
  __shared__ __align__(16) unsigned short Bsm[128 * BS];
  __shared__ int ls_tok[128];

  const int tid = threadIdx.x;
  const int lane = tid & 31, wid = tid >> 5;
  const int wm = wid >> 1, wn = wid & 1;
  const int l16 = lane & 15, hb = lane >> 4;

  if (tid < 128) {
    int r = mt * 128 + tid;
    ls_tok[tid] = list[off + (r < cnt ? r : cnt - 1)];
  }

  const float* wbase = w2 + (size_t)e * IDIM * HDIM;
  const int arow = tid >> 1, ach = (tid & 1) * 16;
  const int bk = (tid >> 5) * 4;
  const int bn = (tid & 31) * 4;

  // per-thread source row for the A async copy (clamped -> always in-bounds)
  int arl = mt * 128 + arow;
  const int gmr = off + (arl < cnt ? arl : cnt - 1);
  const unsigned short* gsrc0 = hbuf + (size_t)gmr * IDIM + ach;
  const unsigned lds_dst = lds_off32(&Asm[arow * AS + ach]);

  v8f acc[2][4];
#pragma unroll
  for (int i = 0; i < 2; ++i)
#pragma unroll
    for (int j = 0; j < 4; ++j) acc[i][j] = (v8f){};

  for (int k0 = 0; k0 < IDIM; k0 += 32) {
    __syncthreads();
    { // stage A: raw bf16 rows via async DMA global->LDS (IOFFSET hits both sides)
      const unsigned short* gsrc = gsrc0 + k0;
      asm volatile("global_load_async_to_lds_b128 %0, %1, off"
                   :: "v"(lds_dst), "v"((unsigned long long)gsrc)
                   : "memory");
      asm volatile("global_load_async_to_lds_b128 %0, %1, off offset:16"
                   :: "v"(lds_dst), "v"((unsigned long long)gsrc)
                   : "memory");
    }
    { // stage B: 4 w2 k-rows x 4 n-cols, pack K-quads -> bf16 transposed [N][K]
      const float* r0 = wbase + (size_t)(k0 + bk) * HDIM + n0 + bn;
      float4 p0 = *(const float4*)(r0);
      float4 p1 = *(const float4*)(r0 + HDIM);
      float4 p2 = *(const float4*)(r0 + 2 * HDIM);
      float4 p3 = *(const float4*)(r0 + 3 * HDIM);
      uint2 q;
      q.x = pk_bf2(p0.x, p1.x); q.y = pk_bf2(p2.x, p3.x);
      *(uint2*)&Bsm[(bn + 0) * BS + bk] = q;
      q.x = pk_bf2(p0.y, p1.y); q.y = pk_bf2(p2.y, p3.y);
      *(uint2*)&Bsm[(bn + 1) * BS + bk] = q;
      q.x = pk_bf2(p0.z, p1.z); q.y = pk_bf2(p2.z, p3.z);
      *(uint2*)&Bsm[(bn + 2) * BS + bk] = q;
      q.x = pk_bf2(p0.w, p1.w); q.y = pk_bf2(p2.w, p3.w);
      *(uint2*)&Bsm[(bn + 3) * BS + bk] = q;
      if (k0 + 32 < IDIM)
        __builtin_prefetch(r0 + (size_t)32 * HDIM, 0, 0);
    }
    asm volatile("s_wait_asynccnt 0x0" ::: "memory");
    __syncthreads();
    const unsigned short* pa = &Asm[(wm * 32 + l16) * AS + hb * 8];
    v16bf a0 = cat16(*(const v8bf*)pa, *(const v8bf*)(pa + 16));
    v16bf a1 = cat16(*(const v8bf*)(pa + 16 * AS), *(const v8bf*)(pa + 16 * AS + 16));
    const unsigned short* pb = &Bsm[(wn * 64 + l16) * BS + hb * 8];
    v16bf bf[4];
#pragma unroll
    for (int j = 0; j < 4; ++j)
      bf[j] = cat16(*(const v8bf*)(pb + j * 16 * BS), *(const v8bf*)(pb + j * 16 * BS + 16));
#pragma unroll
    for (int j = 0; j < 4; ++j) {
      acc[0][j] = __builtin_amdgcn_wmma_f32_16x16x32_bf16(false, a0, false, bf[j], (short)0, acc[0][j], false, false);
      acc[1][j] = __builtin_amdgcn_wmma_f32_16x16x32_bf16(false, a1, false, bf[j], (short)0, acc[1][j], false, false);
    }
  }

#pragma unroll
  for (int j = 0; j < 4; ++j) {
    const int nc = n0 + wn * 64 + j * 16 + l16;
    const float bias = b2[(size_t)e * HDIM + nc];
#pragma unroll
    for (int i = 0; i < 2; ++i) {
#pragma unroll
      for (int r = 0; r < 8; ++r) {
        int m = wm * 32 + i * 16 + r + hb * 8;
        if (mt * 128 + m < cnt) {
          size_t row = (size_t)ls_tok[m] * HDIM;
          out[row + nc] = acc[i][j][r] + bias;
        }
      }
    }
  }
}

// ---------------------------------------------------------------- launch ----
extern "C" void kernel_launch(void* const* d_in, const int* in_sizes, int n_in,
                              void* d_out, int out_size, void* d_ws, size_t ws_size,
                              hipStream_t stream) {
  const float* x   = (const float*)d_in[0];
  const float* rw  = (const float*)d_in[1];
  const float* rb  = (const float*)d_in[2];
  const float* w1  = (const float*)d_in[3];
  const float* b1  = (const float*)d_in[4];
  const float* w2  = (const float*)d_in[5];
  const float* b2  = (const float*)d_in[6];
  const float* emb = (const float*)d_in[7];
  float* out = (float*)d_out;

  char* ws = (char*)d_ws;
  int* eidx    = (int*)(ws);                 // 2048 ints
  int* counts  = (int*)(ws + 8192);          // 8 ints
  int* offsets = (int*)(ws + 8192 + 32);     // 8 ints
  int* cursors = (int*)(ws + 8192 + 64);     // 8 ints
  int* list    = (int*)(ws + 8192 + 128);    // 2048 ints
  unsigned short* hbuf = (unsigned short*)(ws + 32768); // 2048*4096 bf16 = 16 MB

  hipLaunchKernelGGL(k_init,   dim3(1),         dim3(256), 0, stream, counts, cursors);
  hipLaunchKernelGGL(k_router, dim3(TKN / 8),   dim3(256), 0, stream, x, rw, rb, eidx, counts);
  hipLaunchKernelGGL(k_scan,   dim3(1),         dim3(1),   0, stream, counts, offsets);
  hipLaunchKernelGGL(k_fill,   dim3(TKN / 256), dim3(256), 0, stream, eidx, offsets, cursors, list);
  hipLaunchKernelGGL(k_ffn1, dim3(IDIM / 128, TKN / 128, EDIM), dim3(256), 0, stream,
                     x, w1, b1, emb, list, counts, offsets, hbuf);
  hipLaunchKernelGGL(k_ffn2, dim3(HDIM / 128, TKN / 128, EDIM), dim3(256), 0, stream,
                     hbuf, w2, b2, list, counts, offsets, out);
}
